// Encoder_70927089926824
// MI455X (gfx1250) — compile-verified, matrix operand order checked by reading, not run on hardware
//
#include <hip/hip_runtime.h>

#define B_    8
#define S_    1024
#define HID_  1024
#define NH_   16
#define HD_   64
#define D_    4
#define F_    4096
#define L_    4
#define M_    (B_*S_)          /* 8192 tokens */
#define EPS_  0.1f
#define LNE_  1e-5f

typedef unsigned int  u32;
typedef unsigned short u16;
typedef __attribute__((ext_vector_type(16))) __bf16 v16bf;
typedef __attribute__((ext_vector_type(8)))  float  v8f;
typedef __attribute__((ext_vector_type(4)))  u32    u32x4;
typedef __attribute__((ext_vector_type(2)))  u32    u32x2;

union V16U { v16bf v; u32x4 u[2]; };

__device__ __forceinline__ u16 bfbits(float f) {
  union { float f; u32 u; } x; x.f = f;
  u32 r = x.u + 0x7fffu + ((x.u >> 16) & 1u);   // round-to-nearest-even
  return (u16)(r >> 16);
}
__device__ __forceinline__ u32 pk2(float a, float b) {
  return (u32)bfbits(a) | ((u32)bfbits(b) << 16);
}

// Butterfly exchange via ds_swizzle_b32 (group-of-32, immediate xor mask).
#define XOR_SWZ(v, m) \
  __int_as_float(__builtin_amdgcn_ds_swizzle(__float_as_int(v), ((m) << 10) | 0x1F))

// Async DMA: copy 16 bytes global -> LDS without a VGPR round trip.
// VDST holds the LDS byte offset (low 32 bits of the generic shared pointer,
// as the shared aperture maps it to LDS); GV mode, saddr = off.
// Tracked by ASYNCcnt; completion enforced with s_wait_asynccnt.
__device__ __forceinline__ void async_b128(const u16* lds_dst, const u16* g_src) {
  const u32 lds_off = (u32)(size_t)(const void*)lds_dst;
  asm volatile("global_load_async_to_lds_b128 %0, %1, off"
               :: "v"(lds_off), "v"(g_src) : "memory");
}
__device__ __forceinline__ void wait_async0() {
  asm volatile("s_wait_asynccnt 0x0" ::: "memory");
}

// Load one 16x32 bf16 fragment (A-layout per CDNA5 ISA 7.12.2) from a
// row-major tile at `p` with row stride `stride` (elements).
__device__ __forceinline__ v16bf load_frag(const u16* p, int stride) {
  const int lane = threadIdx.x & 31;
  const u16* r = p + (lane & 15) * stride + ((lane >> 4) << 3);
  V16U out;
  out.u[0] = *(const u32x4*)(r);
  out.u[1] = *(const u32x4*)(r + 16);
  return out.v;
}

// ---------------------------------------------------------------------------
// GEMM: C[M,N] = A[M,K](bf16,row-major) x Bt[N,K](bf16,row-major == B^T)
// 128x128x32 block tile, 8 waves, each wave 64x32 via 4x2 WMMA frags.
// Double-buffered LDS fed by global_load_async_to_lds_b128: the async DMA of
// tile t+1 overlaps the WMMA burst of tile t; s_wait_asynccnt + barrier gate
// each buffer swap.
// ---------------------------------------------------------------------------
#define BM 128
#define BN 128
#define BKK 32
#define LS 40   /* padded LDS row stride (elements); 80B keeps 16B alignment */

enum { EPI_GATED_BF16 = 0, EPI_RELU_BF16 = 1, EPI_BIAS_F32 = 2 };

template<int EPI>
__global__ __launch_bounds__(256) void gemm_bf16_kernel(
    const u16* __restrict__ A, const u16* __restrict__ Bt,
    int M, int N, int K,
    const float* __restrict__ bias,   // [N] or [4*N] for gated epilogue
    const float* __restrict__ gates,  // [M,4] (already scaled by 1/D) for gated
    float* __restrict__ Cf, u16* __restrict__ Cb)
{
  __shared__ u16 As[2][BM * LS];
  __shared__ u16 Bs[2][BN * LS];
  const int tid  = threadIdx.x;
  const int lane = tid & 31;
  const int wave = tid >> 5;
  const int bm = blockIdx.x * BM;
  const int bn = blockIdx.y * BN;
  const int wm = (wave & 1) * 64;
  const int wn = (wave >> 1) * 32;

  const int lrow = tid >> 1;          // 0..127
  const int lcol = (tid & 1) * 16;    // 0 or 16
  const u16* Ag = A  + (size_t)(bm + lrow) * K + lcol;
  const u16* Bg = Bt + (size_t)(bn + lrow) * K + lcol;
  const int woff = lrow * LS + lcol;

  v8f z = {0.f,0.f,0.f,0.f,0.f,0.f,0.f,0.f};
  v8f acc[4][2];
#pragma unroll
  for (int i = 0; i < 4; ++i)
#pragma unroll
    for (int j = 0; j < 2; ++j) acc[i][j] = z;

  const int nt = K / BKK;
  // prologue: async DMA tile 0 -> buffer 0
  async_b128(As[0] + woff,     Ag);
  async_b128(As[0] + woff + 8, Ag + 8);
  async_b128(Bs[0] + woff,     Bg);
  async_b128(Bs[0] + woff + 8, Bg + 8);

  for (int t = 0; t < nt; ++t) {
    wait_async0();                 // my async writes for tile t are in LDS
    __syncthreads();               // everyone's are
    const int cur = t & 1;
    const int nxt = cur ^ 1;
    if (t + 1 < nt) {              // DMA tile t+1 while tile t computes
      const int kb = (t + 1) * BKK;
      async_b128(As[nxt] + woff,     Ag + kb);
      async_b128(As[nxt] + woff + 8, Ag + kb + 8);
      async_b128(Bs[nxt] + woff,     Bg + kb);
      async_b128(Bs[nxt] + woff + 8, Bg + kb + 8);
      if (t + 2 < nt) {            // global_prefetch_b8 for tile t+2
        __builtin_prefetch(Ag + kb + BKK, 0, 0);
        __builtin_prefetch(Bg + kb + BKK, 0, 0);
      }
    }

    v16bf af[4], bfr[2];
#pragma unroll
    for (int i = 0; i < 4; ++i) af[i]  = load_frag(As[cur] + (wm + i*16) * LS, LS);
#pragma unroll
    for (int j = 0; j < 2; ++j) bfr[j] = load_frag(Bs[cur] + (wn + j*16) * LS, LS);
#pragma unroll
    for (int i = 0; i < 4; ++i)
#pragma unroll
      for (int j = 0; j < 2; ++j)
        acc[i][j] = __builtin_amdgcn_wmma_f32_16x16x32_bf16(
            false, af[i], false, bfr[j], (short)0, acc[i][j], false, false);
  }

  const int hi8 = (lane >> 4) << 3;
  const int cn  = lane & 15;
#pragma unroll
  for (int i = 0; i < 4; ++i) {
#pragma unroll
    for (int r = 0; r < 8; ++r) {
      const int grow = bm + wm + i*16 + hi8 + r;
      if constexpr (EPI == EPI_GATED_BF16) {
        const float g0 = gates[grow*4+0], gg1 = gates[grow*4+1];
        const float g2 = gates[grow*4+2], g3  = gates[grow*4+3];
#pragma unroll
        for (int j = 0; j < 2; ++j) {
          const int gcol = bn + wn + j*16 + cn;
          const float bvv = g0*bias[gcol] + gg1*bias[N+gcol]
                          + g2*bias[2*N+gcol] + g3*bias[3*N+gcol];
          Cb[(size_t)grow * N + gcol] = bfbits(acc[i][j][r] + bvv);
        }
      } else if constexpr (EPI == EPI_RELU_BF16) {
#pragma unroll
        for (int j = 0; j < 2; ++j) {
          const int gcol = bn + wn + j*16 + cn;
          Cb[(size_t)grow * N + gcol] = bfbits(fmaxf(acc[i][j][r] + bias[gcol], 0.f));
        }
      } else {
#pragma unroll
        for (int j = 0; j < 2; ++j) {
          const int gcol = bn + wn + j*16 + cn;
          Cf[(size_t)grow * N + gcol] = acc[i][j][r] + bias[gcol];
        }
      }
    }
  }
}

// ---------------------------------------------------------------------------
// Flash attention: grid.x = B*NH, grid.y = S/64. 4 waves/block, 16 q-rows/wave.
// Q,K,V,O are bf16 [B*S, HID]; head h occupies columns h*64..h*64+63.
// ---------------------------------------------------------------------------
__global__ __launch_bounds__(128) void attn_kernel(
    const u16* __restrict__ Q, const u16* __restrict__ Km,
    const u16* __restrict__ V, u16* __restrict__ O)
{
  __shared__ u16 Ks[32 * 72];       // K tile [32 keys][64 dims], stride 72
  __shared__ u16 Vt[64 * 40];       // V^T tile [64 dims][32 keys], stride 40
  __shared__ u16 Ps[4 * 16 * 40];   // per-wave probability tile [16 q][32 keys]
  const int tid = threadIdx.x;
  const int lane = tid & 31;
  const int wave = tid >> 5;
  const int b = blockIdx.x >> 4;
  const int h = blockIdx.x & 15;
  const int q0 = blockIdx.y * 64 + wave * 16;
  const size_t hoff = (size_t)h * HD_;

  const u16* Qb = Q + (size_t)(b * S_ + q0) * HID_ + hoff;
  const v16bf qa0 = load_frag(Qb,      HID_);   // q dims 0..31
  const v16bf qa1 = load_frag(Qb + 32, HID_);   // q dims 32..63

  const int lrow = tid >> 2;          // 0..31 (key within tile)
  const int lcol = (tid & 3) * 16;    // 0/16/32/48 (dim)
  const u16* Kg = Km + ((size_t)(b * S_) + lrow) * HID_ + hoff + lcol;
  const u16* Vg = V  + ((size_t)(b * S_) + lrow) * HID_ + hoff + lcol;

  v8f z = {0.f,0.f,0.f,0.f,0.f,0.f,0.f,0.f};
  v8f o[4];
#pragma unroll
  for (int j = 0; j < 4; ++j) o[j] = z;
  float mo[8], ls[8];
#pragma unroll
  for (int r = 0; r < 8; ++r) { mo[r] = -1e30f; ls[r] = 0.f; }
  u16* Pw = Ps + wave * 16 * 40;
  const int hi8 = (lane >> 4) << 3;
  const int cn  = lane & 15;

  for (int kt = 0; kt < S_; kt += 32) {
    const u32x4 k0 = *(const u32x4*)(Kg + (size_t)kt * HID_);
    const u32x4 k1 = *(const u32x4*)(Kg + (size_t)kt * HID_ + 8);
    const u32x4 v0 = *(const u32x4*)(Vg + (size_t)kt * HID_);
    const u32x4 v1 = *(const u32x4*)(Vg + (size_t)kt * HID_ + 8);
    *(u32x4*)(Ks + lrow*72 + lcol)     = k0;
    *(u32x4*)(Ks + lrow*72 + lcol + 8) = k1;
    union { u32x4 u[2]; u16 s[16]; } vv; vv.u[0] = v0; vv.u[1] = v1;
#pragma unroll
    for (int e = 0; e < 16; ++e) Vt[(lcol + e)*40 + lrow] = vv.s[e];
    __syncthreads();

    v8f sc[2];
#pragma unroll
    for (int jn = 0; jn < 2; ++jn) {
      const v16bf bk0 = load_frag(Ks + jn*16*72,      72);
      const v16bf bk1 = load_frag(Ks + jn*16*72 + 32, 72);
      v8f s = z;
      s = __builtin_amdgcn_wmma_f32_16x16x32_bf16(false, qa0, false, bk0, (short)0, s, false, false);
      s = __builtin_amdgcn_wmma_f32_16x16x32_bf16(false, qa1, false, bk1, (short)0, s, false, false);
      sc[jn] = s;
    }
#pragma unroll
    for (int r = 0; r < 8; ++r) {
      const float s0 = sc[0][r] * 0.125f;     // 1/sqrt(64)
      const float s1 = sc[1][r] * 0.125f;
      float mx = fmaxf(s0, s1);
      mx = fmaxf(mx, XOR_SWZ(mx, 1));
      mx = fmaxf(mx, XOR_SWZ(mx, 2));
      mx = fmaxf(mx, XOR_SWZ(mx, 4));
      mx = fmaxf(mx, XOR_SWZ(mx, 8));
      const float mn = fmaxf(mo[r], mx);
      const float p0 = __expf(s0 - mn), p1 = __expf(s1 - mn);
      float psum = p0 + p1;
      psum += XOR_SWZ(psum, 1);
      psum += XOR_SWZ(psum, 2);
      psum += XOR_SWZ(psum, 4);
      psum += XOR_SWZ(psum, 8);
      const float corr = __expf(mo[r] - mn);
      ls[r] = ls[r] * corr + psum;
      mo[r] = mn;
#pragma unroll
      for (int j = 0; j < 4; ++j) o[j][r] *= corr;
      const int prow = hi8 + r;
      Pw[prow*40 + cn]      = bfbits(p0);
      Pw[prow*40 + 16 + cn] = bfbits(p1);
    }
    const v16bf pa = load_frag(Pw, 40);
#pragma unroll
    for (int j = 0; j < 4; ++j) {
      const v16bf bv = load_frag(Vt + j*16*40, 40);
      o[j] = __builtin_amdgcn_wmma_f32_16x16x32_bf16(false, pa, false, bv, (short)0, o[j], false, false);
    }
    __syncthreads();
  }
#pragma unroll
  for (int j = 0; j < 4; ++j)
#pragma unroll
    for (int r = 0; r < 8; ++r) {
      const int row = q0 + hi8 + r;
      O[(size_t)(b * S_ + row) * HID_ + hoff + j*16 + cn] = bfbits(o[j][r] / ls[r]);
    }
}

// ---------------------------------------------------------------------------
// Gate heads: one warp per token; logits = x @ Wr + br; softmax over D=4.
// gq/gk/gv stored pre-scaled by 1/D for the folded mix-GEMM.
// ---------------------------------------------------------------------------
__global__ __launch_bounds__(256) void gate_kernel(
    const float* __restrict__ x,
    const float* __restrict__ Wrq, const float* __restrict__ brq,
    const float* __restrict__ Wrk, const float* __restrict__ brk,
    const float* __restrict__ Wrv, const float* __restrict__ brv,
    float* __restrict__ gq, float* __restrict__ gk, float* __restrict__ gv,
    float* __restrict__ dom)
{
  const int m = blockIdx.x * 8 + (threadIdx.x >> 5);
  const int lane = threadIdx.x & 31;
  const float* xr = x + (size_t)m * HID_;
  float acc[3][4];
#pragma unroll
  for (int g = 0; g < 3; ++g)
#pragma unroll
    for (int d = 0; d < 4; ++d) acc[g][d] = 0.f;
  for (int hh = lane; hh < HID_; hh += 32) {
    const float xv = xr[hh];
    const float4 wq = *(const float4*)(Wrq + hh*4);
    const float4 wk = *(const float4*)(Wrk + hh*4);
    const float4 wv = *(const float4*)(Wrv + hh*4);
    acc[0][0] += xv*wq.x; acc[0][1] += xv*wq.y; acc[0][2] += xv*wq.z; acc[0][3] += xv*wq.w;
    acc[1][0] += xv*wk.x; acc[1][1] += xv*wk.y; acc[1][2] += xv*wk.z; acc[1][3] += xv*wk.w;
    acc[2][0] += xv*wv.x; acc[2][1] += xv*wv.y; acc[2][2] += xv*wv.z; acc[2][3] += xv*wv.w;
  }
#pragma unroll
  for (int g = 0; g < 3; ++g)
#pragma unroll
    for (int d = 0; d < 4; ++d) {
      float v = acc[g][d];
      v += XOR_SWZ(v, 1);
      v += XOR_SWZ(v, 2);
      v += XOR_SWZ(v, 4);
      v += XOR_SWZ(v, 8);
      v += XOR_SWZ(v, 16);
      acc[g][d] = v;
    }
  if (lane == 0) {
    const float* bs[3] = { brq, brk, brv };
    float sm[3][4];
#pragma unroll
    for (int g = 0; g < 3; ++g) {
      float lg[4], mx = -1e30f;
#pragma unroll
      for (int d = 0; d < 4; ++d) { lg[d] = acc[g][d] + bs[g][d]; mx = fmaxf(mx, lg[d]); }
      float s = 0.f;
#pragma unroll
      for (int d = 0; d < 4; ++d) { lg[d] = __expf(lg[d] - mx); s += lg[d]; }
#pragma unroll
      for (int d = 0; d < 4; ++d) sm[g][d] = (1.f - EPS_) * lg[d] / s + EPS_ / (float)D_;
    }
    float* outs[3] = { gq, gk, gv };
#pragma unroll
    for (int g = 0; g < 3; ++g)
#pragma unroll
      for (int d = 0; d < 4; ++d)
        outs[g][m*4 + d] = sm[g][d] * 0.25f;     // fold 1/D
    if (dom)
#pragma unroll
      for (int d = 0; d < 4; ++d)
        dom[m*4 + d] = (sm[0][d] + sm[1][d] + sm[2][d]) * (1.f/3.f);
  }
}

// A'[m, d*HID+h] = g[m,d] * x[m,h]  (bf16), g pre-scaled by 1/D.
__global__ __launch_bounds__(256) void aprime_kernel(
    const float* __restrict__ x, const float* __restrict__ g, u16* __restrict__ Ap)
{
  const size_t i = ((size_t)blockIdx.x * 256 + threadIdx.x) * 4;
  const int m  = (int)(i >> 12);
  const int c  = (int)(i & 4095);
  const int d  = c >> 10;
  const int hb = c & 1023;
  const float gg = g[m*4 + d];
  const float4 xv = *(const float4*)(x + (size_t)m * HID_ + hb);
  u32x2 out;
  out[0] = pk2(gg*xv.x, gg*xv.y);
  out[1] = pk2(gg*xv.z, gg*xv.w);
  *(u32x2*)(Ap + i) = out;
}

// Wt[n*K + k] = bf16(W[k*N + n])  — transpose + convert weights for GEMM.
__global__ __launch_bounds__(256) void wconv_kernel(
    const float* __restrict__ W, u16* __restrict__ Wt, int K, int N)
{
  const size_t i = ((size_t)blockIdx.x * 256 + threadIdx.x) * 4;
  const int n  = (int)(i / K);
  const int k0 = (int)(i % K);
  const float a = W[(size_t)(k0+0)*N + n];
  const float b = W[(size_t)(k0+1)*N + n];
  const float c = W[(size_t)(k0+2)*N + n];
  const float d = W[(size_t)(k0+3)*N + n];
  u32x2 out; out[0] = pk2(a, b); out[1] = pk2(c, d);
  *(u32x2*)(Wt + i) = out;
}

// out = LN(resid + y) * gamma + beta ; writes f32 and bf16 copies.
__global__ __launch_bounds__(256) void ln_kernel(
    const float* __restrict__ resid, const float* __restrict__ y,
    const float* __restrict__ gam, const float* __restrict__ bet,
    float* __restrict__ of, u16* __restrict__ ob)
{
  __shared__ float red[8];
  const int row = blockIdx.x;
  const int tid = threadIdx.x;
  const int lane = tid & 31, wv = tid >> 5;
  const size_t base = (size_t)row * HID_ + tid * 4;
  const float4 rv = *(const float4*)(resid + base);
  const float4 yv = *(const float4*)(y + base);
  float v[4] = { rv.x+yv.x, rv.y+yv.y, rv.z+yv.z, rv.w+yv.w };
  float s = v[0]+v[1]+v[2]+v[3];
  s += XOR_SWZ(s, 1);
  s += XOR_SWZ(s, 2);
  s += XOR_SWZ(s, 4);
  s += XOR_SWZ(s, 8);
  s += XOR_SWZ(s, 16);
  if (lane == 0) red[wv] = s;
  __syncthreads();
  float tot = 0.f;
#pragma unroll
  for (int w = 0; w < 8; ++w) tot += red[w];
  const float mean = tot * (1.f/(float)HID_);
  __syncthreads();
  float vs = 0.f;
#pragma unroll
  for (int i = 0; i < 4; ++i) { const float dd = v[i]-mean; vs += dd*dd; }
  vs += XOR_SWZ(vs, 1);
  vs += XOR_SWZ(vs, 2);
  vs += XOR_SWZ(vs, 4);
  vs += XOR_SWZ(vs, 8);
  vs += XOR_SWZ(vs, 16);
  if (lane == 0) red[wv] = vs;
  __syncthreads();
  float vtot = 0.f;
#pragma unroll
  for (int w = 0; w < 8; ++w) vtot += red[w];
  const float inv = rsqrtf(vtot * (1.f/(float)HID_) + LNE_);
  const int c0 = tid * 4;
  float o0 = (v[0]-mean)*inv*gam[c0+0] + bet[c0+0];
  float o1 = (v[1]-mean)*inv*gam[c0+1] + bet[c0+1];
  float o2 = (v[2]-mean)*inv*gam[c0+2] + bet[c0+2];
  float o3 = (v[3]-mean)*inv*gam[c0+3] + bet[c0+3];
  *(float4*)(of + base) = make_float4(o0, o1, o2, o3);
  u32x2 ob2; ob2[0] = pk2(o0, o1); ob2[1] = pk2(o2, o3);
  *(u32x2*)(ob + base) = ob2;
}

// x[m,h] = tok_emb[src[m],h]*sqrt(HID) + pos_emb[m%S, h]
__global__ __launch_bounds__(256) void embed_kernel(
    const int* __restrict__ src, const float* __restrict__ tok,
    const float* __restrict__ pos, float* __restrict__ x)
{
  const size_t i = ((size_t)blockIdx.x * 256 + threadIdx.x) * 4;
  const int m  = (int)(i >> 10);
  const int hh = (int)(i & 1023);
  const int s  = m & (S_ - 1);
  const int t  = src[m];
  const float4 tv = *(const float4*)(tok + (size_t)t * HID_ + hh);
  const float4 pv = *(const float4*)(pos + (size_t)s * HID_ + hh);
  *(float4*)(x + i) = make_float4(tv.x*32.f+pv.x, tv.y*32.f+pv.y,
                                  tv.z*32.f+pv.z, tv.w*32.f+pv.w);
}

// ---------------------------------------------------------------------------
extern "C" void kernel_launch(void* const* d_in, const int* in_sizes, int n_in,
                              void* d_out, int out_size, void* d_ws, size_t ws_size,
                              hipStream_t stream) {
  (void)in_sizes; (void)n_in; (void)out_size; (void)ws_size;
  const int*   src = (const int*)  d_in[0];
  const float* tok = (const float*)d_in[2];
  const float* pos = (const float*)d_in[3];
  const float* Wrq = (const float*)d_in[4];  const float* brq = (const float*)d_in[5];
  const float* Wrk = (const float*)d_in[6];  const float* brk = (const float*)d_in[7];
  const float* Wrv = (const float*)d_in[8];  const float* brv = (const float*)d_in[9];
  const float* Wq  = (const float*)d_in[10]; const float* bq  = (const float*)d_in[11];
  const float* Wk  = (const float*)d_in[12]; const float* bk  = (const float*)d_in[13];
  const float* Wv  = (const float*)d_in[14]; const float* bv  = (const float*)d_in[15];
  const float* Wo  = (const float*)d_in[16]; const float* bo  = (const float*)d_in[17];
  const float* g1  = (const float*)d_in[18]; const float* b1  = (const float*)d_in[19];
  const float* Wf1 = (const float*)d_in[20]; const float* bf1 = (const float*)d_in[21];
  const float* Wf2 = (const float*)d_in[22]; const float* bf2 = (const float*)d_in[23];
  const float* g2  = (const float*)d_in[24]; const float* b2  = (const float*)d_in[25];

  float* OUTX = (float*)d_out;
  float* OUTD = OUTX + (size_t)M_ * HID_;

  char* w = (char*)d_ws;
  auto carve = [&](size_t bytes) -> char* {
    char* p = w; w += (bytes + 255) & ~(size_t)255; return p;
  };
  float* X    = (float*)carve((size_t)M_ * HID_ * 4);
  float* X1   = (float*)carve((size_t)M_ * HID_ * 4);
  float* Y    = (float*)carve((size_t)M_ * HID_ * 4);
  u16*   XB   = (u16*)  carve((size_t)M_ * HID_ * 2);
  u16*   QB   = (u16*)  carve((size_t)M_ * HID_ * 2);
  u16*   KB   = (u16*)  carve((size_t)M_ * HID_ * 2);
  u16*   VB   = (u16*)  carve((size_t)M_ * HID_ * 2);
  u16*   CTXB = (u16*)  carve((size_t)M_ * HID_ * 2);
  u16*   AP   = (u16*)  carve((size_t)M_ * 4096 * 2);   // A' / FFN hidden
  u16*   WT   = (u16*)  carve((size_t)4096 * 1024 * 2); // staged bf16 weights
  float* GQ   = (float*)carve((size_t)M_ * 4 * 4);
  float* GK   = (float*)carve((size_t)M_ * 4 * 4);
  float* GV   = (float*)carve((size_t)M_ * 4 * 4);

  embed_kernel<<<8192, 256, 0, stream>>>(src, tok, pos, X);

  for (int l = 0; l < L_; ++l) {
    const float* Wq_l  = Wq  + (size_t)l * D_ * HID_ * HID_;
    const float* Wk_l  = Wk  + (size_t)l * D_ * HID_ * HID_;
    const float* Wv_l  = Wv  + (size_t)l * D_ * HID_ * HID_;
    const float* Wo_l  = Wo  + (size_t)l * HID_ * HID_;
    const float* Wf1_l = Wf1 + (size_t)l * HID_ * F_;
    const float* Wf2_l = Wf2 + (size_t)l * F_ * HID_;

    gate_kernel<<<M_/8, 256, 0, stream>>>(
        X, Wrq + (size_t)l*HID_*D_, brq + l*D_,
           Wrk + (size_t)l*HID_*D_, brk + l*D_,
           Wrv + (size_t)l*HID_*D_, brv + l*D_,
        GQ, GK, GV, (l == L_-1) ? OUTD : nullptr);

    // Q = A'q @ [Wq_0;..;Wq_3]  (+ gated bias)
    wconv_kernel<<<4096, 256, 0, stream>>>(Wq_l, WT, 4096, 1024);
    aprime_kernel<<<32768, 256, 0, stream>>>(X, GQ, AP);
    gemm_bf16_kernel<EPI_GATED_BF16><<<dim3(64, 8), 256, 0, stream>>>(
        AP, WT, M_, 1024, 4096, bq + (size_t)l*D_*HID_, GQ, nullptr, QB);
    // K
    wconv_kernel<<<4096, 256, 0, stream>>>(Wk_l, WT, 4096, 1024);
    aprime_kernel<<<32768, 256, 0, stream>>>(X, GK, AP);
    gemm_bf16_kernel<EPI_GATED_BF16><<<dim3(64, 8), 256, 0, stream>>>(
        AP, WT, M_, 1024, 4096, bk + (size_t)l*D_*HID_, GK, nullptr, KB);
    // V
    wconv_kernel<<<4096, 256, 0, stream>>>(Wv_l, WT, 4096, 1024);
    aprime_kernel<<<32768, 256, 0, stream>>>(X, GV, AP);
    gemm_bf16_kernel<EPI_GATED_BF16><<<dim3(64, 8), 256, 0, stream>>>(
        AP, WT, M_, 1024, 4096, bv + (size_t)l*D_*HID_, GV, nullptr, VB);

    attn_kernel<<<dim3(B_*NH_, S_/64), 128, 0, stream>>>(QB, KB, VB, CTXB);

    // Wo + bias (f32), residual+LN1
    wconv_kernel<<<1024, 256, 0, stream>>>(Wo_l, WT, 1024, 1024);
    gemm_bf16_kernel<EPI_BIAS_F32><<<dim3(64, 8), 256, 0, stream>>>(
        CTXB, WT, M_, 1024, 1024, bo + (size_t)l*HID_, nullptr, Y, nullptr);
    ln_kernel<<<M_, 256, 0, stream>>>(X, Y, g1 + (size_t)l*HID_, b1 + (size_t)l*HID_, X1, XB);

    // FFN
    wconv_kernel<<<4096, 256, 0, stream>>>(Wf1_l, WT, 1024, 4096);
    gemm_bf16_kernel<EPI_RELU_BF16><<<dim3(64, 32), 256, 0, stream>>>(
        XB, WT, M_, 4096, 1024, bf1 + (size_t)l*F_, nullptr, nullptr, AP);
    wconv_kernel<<<4096, 256, 0, stream>>>(Wf2_l, WT, 4096, 1024);
    gemm_bf16_kernel<EPI_BIAS_F32><<<dim3(64, 8), 256, 0, stream>>>(
        AP, WT, M_, 1024, 4096, bf2 + (size_t)l*HID_, nullptr, Y, nullptr);
    ln_kernel<<<M_, 256, 0, stream>>>(X1, Y, g2 + (size_t)l*HID_, b2 + (size_t)l*HID_,
                                      (l == L_-1) ? OUTX : X, XB);
  }
}
